// GATNet_7052336300583
// MI455X (gfx1250) — matrix operand order repaired and checked by reading, not compiled, not run on hardware
//
#include <hip/hip_runtime.h>
#include <hip/hip_bf16.h>
#include <stdint.h>

typedef _Float16 v16h __attribute__((ext_vector_type(16)));
typedef float    v8f  __attribute__((ext_vector_type(8)));

union FragH {
    v16h     v;
    uint32_t u[8];
    _Float16 h[16];
    uint4    q[2];
};

__device__ __forceinline__ unsigned f2key(float f) {
    unsigned u = __float_as_uint(f);
    return (u & 0x80000000u) ? ~u : (u ^ 0x80000000u);
}
__device__ __forceinline__ float key2f(unsigned k) {
    unsigned u = (k & 0x80000000u) ? (k ^ 0x80000000u) : ~k;
    return __uint_as_float(u);
}

// ---------------- f32 -> f16 conversion, 4 elements/thread ----------------
__global__ void cvt_f32_to_f16(const float* __restrict__ in, _Float16* __restrict__ out,
                               long long n4) {   // n4 = n/4
    long long i = (long long)blockIdx.x * blockDim.x + threadIdx.x;
    long long stride = (long long)gridDim.x * blockDim.x;
    for (; i < n4; i += stride) {
        float4 f = ((const float4*)in)[i];
        union { _Float16 h[4]; uint2 u; } o;
        o.h[0] = (_Float16)f.x; o.h[1] = (_Float16)f.y;
        o.h[2] = (_Float16)f.z; o.h[3] = (_Float16)f.w;
        ((uint2*)out)[i] = o.u;
    }
}

// ---------------- pack W (f32, K x F row-major) into per-lane WMMA B fragments ----------
// Bpack layout: [(ct*(K/32) + kci)*32 + lane] -> 2 x uint4 (16 halves, lane-contiguous).
// Fragment layout (contiguous k order, matching the A loads in gemm_wmma): lane l (n=l&15, half=l>>4), dword v holds
// rows kb = kci*32 + 2v + 16*half and kb+1 of column n.
__global__ void pack_b_kernel(const float* __restrict__ W, uint4* __restrict__ Bpack,
                              int K, int F) {
    int t = blockIdx.x * blockDim.x + threadIdx.x;
    int kch = K >> 5;
    int total = (F >> 4) * kch * 32;
    if (t >= total) return;
    int lane = t & 31;
    int kci  = (t >> 5) % kch;
    int ct   = t / (32 * kch);
    int nl = lane & 15, half = lane >> 4;
    int col = ct * 16 + nl;
    FragH fb;
#pragma unroll
    for (int v = 0; v < 8; ++v) {
        int kb = kci * 32 + 2 * v + 16 * half;
        fb.h[2 * v]     = (_Float16)W[(size_t)kb * F + col];
        fb.h[2 * v + 1] = (_Float16)W[(size_t)(kb + 1) * F + col];
    }
    Bpack[2 * t]     = fb.q[0];
    Bpack[2 * t + 1] = fb.q[1];
}

// ---------------- WMMA GEMM: C[N x F] = act(A[N x K] @ B + bias) ----------------
// One wave owns a 16-row strip: A fragments live in registers; the wave sweeps
// all F/16 column tiles reading pre-packed, lane-contiguous B fragments with the
// next chunk's loads issued before the current WMMA (software pipelining).
// N must be a multiple of 16 (100000 = 6250*16): no per-row bounds checks.
template<int K, int F, bool OUT16, bool RELU, bool BIAS>
__global__ __launch_bounds__(128)
void gemm_wmma(const _Float16* __restrict__ A, const uint4* __restrict__ Bpack,
               const float* __restrict__ bias, void* __restrict__ C, int N)
{
    constexpr int KCH = K / 32;
    const int wave = threadIdx.x >> 5;
    const int lane = threadIdx.x & 31;
    const int half = lane >> 4;
    const int nl   = lane & 15;
    const int row0 = (blockIdx.x * 4 + wave) * 16;
    if (row0 >= N) return;                       // wave-uniform exit

    const _Float16* __restrict__ arow = A + (size_t)(row0 + nl) * K;

    FragH fa[KCH];
#pragma unroll
    for (int kci = 0; kci < KCH; ++kci) {
        fa[kci].q[0] = *(const uint4*)(arow + kci * 32 + 16 * half);      // k: +16h+0..+7
        fa[kci].q[1] = *(const uint4*)(arow + kci * 32 + 16 * half + 8);  // k: +16h+8..+15
    }

#pragma unroll 1
    for (int ct = 0; ct < F / 16; ++ct) {
        v8f acc = {};
        const uint4* bp = Bpack + (size_t)(ct * KCH) * 64 + 2 * lane;
        FragH fb, fbn;
        fb.q[0] = bp[0];
        fb.q[1] = bp[1];
#pragma unroll
        for (int kci = 0; kci < KCH; ++kci) {
            if (kci + 1 < KCH) {                 // prefetch next chunk before WMMA
                fbn.q[0] = bp[64 * (kci + 1)];
                fbn.q[1] = bp[64 * (kci + 1) + 1];
            }
            acc = __builtin_amdgcn_wmma_f32_16x16x32_f16(
                false, fa[kci].v, false, fb.v, (short)0, acc, false, false);
            fb = fbn;
        }
        const int col = ct * 16 + nl;
        const float bv = BIAS ? bias[col] : 0.0f;
#pragma unroll
        for (int v = 0; v < 8; ++v) {
            const int row = row0 + v + 8 * half;
            float val = acc[v];
            if (BIAS) val += bv;
            if (RELU) val = fmaxf(val, 0.0f);
            if (OUT16) ((_Float16*)C)[(size_t)row * F + col] = (_Float16)val;
            else       ((float*)C)[(size_t)row * F + col]    = val;
        }
    }
}

// ---------------- a_src / a_dst = h . att  (wave per node) ----------------
__global__ __launch_bounds__(256)
void rowdot2_kernel(const float* __restrict__ h, const float* __restrict__ att_s,
                    const float* __restrict__ att_d, float* __restrict__ asrc,
                    float* __restrict__ adst, int N)
{
    int n = blockIdx.x * 8 + (threadIdx.x >> 5);
    if (n >= N) return;
    int lane = threadIdx.x & 31;
    const float4 hv = ((const float4*)(h + (size_t)n * 128))[lane];
    const float4 sv = ((const float4*)att_s)[lane];
    const float4 dv = ((const float4*)att_d)[lane];
    float s = hv.x * sv.x + hv.y * sv.y + hv.z * sv.z + hv.w * sv.w;
    float d = hv.x * dv.x + hv.y * dv.y + hv.z * dv.z + hv.w * dv.w;
#pragma unroll
    for (int msk = 16; msk; msk >>= 1) {
        s += __shfl_xor(s, msk, 32);
        d += __shfl_xor(d, msk, 32);
    }
    if (lane == 0) { asrc[n] = s; adst[n] = d; }
}

// ---------------- per-node init: segment-max key / Z accumulator ----------------
__global__ void init_nodes_kernel(unsigned* __restrict__ mkey, float* __restrict__ z, int N) {
    int n = blockIdx.x * blockDim.x + threadIdx.x;
    if (n < N) { mkey[n] = 0u; z[n] = 0.0f; }   // key 0 == -NaN == minimum
}

// ---------------- init output accumulator with bias ----------------
__global__ void init_out_kernel(float* __restrict__ out, const float* __restrict__ bias,
                                long long total) {
    long long i = (long long)blockIdx.x * blockDim.x + threadIdx.x;
    long long stride = (long long)gridDim.x * blockDim.x;
    for (; i < total; i += stride) out[i] = bias[i & 127];
}

// ---------------- edge pass 1: logits + segment max ----------------
__global__ void edge_logits_kernel(const int* __restrict__ src, const int* __restrict__ dst,
                                   const float* __restrict__ asrc, const float* __restrict__ adst,
                                   float* __restrict__ eo, unsigned* __restrict__ mkey,
                                   long long E, long long Etot)
{
    long long i = (long long)blockIdx.x * blockDim.x + threadIdx.x;
    if (i >= Etot) return;
    int s, d;
    if (i < E) { s = src[i]; d = dst[i]; } else { s = d = (int)(i - E); }
    float e = asrc[s] + adst[d];
    e = (e > 0.0f) ? e : 0.2f * e;              // leaky_relu, slope 0.2
    eo[i] = e;
    atomicMax(&mkey[d], f2key(e));
}

// ---------------- edge pass 2: p = exp(e - m[dst]); Z += p ----------------
__global__ void edge_exp_kernel(const int* __restrict__ dst, float* __restrict__ ep,
                                const unsigned* __restrict__ mkey, float* __restrict__ z,
                                long long E, long long Etot)
{
    long long i = (long long)blockIdx.x * blockDim.x + threadIdx.x;
    if (i >= Etot) return;
    int d = (i < E) ? dst[i] : (int)(i - E);
    float p = __expf(ep[i] - key2f(mkey[d]));
    ep[i] = p;
    atomicAdd(&z[d], p);
}

// ---------------- edge pass 3: out[dst] += alpha * h[src]  (wave per edge) ----------------
__global__ __launch_bounds__(256)
void edge_aggregate_kernel(const int* __restrict__ src, const int* __restrict__ dst,
                           const float* __restrict__ p, const float* __restrict__ z,
                           const float* __restrict__ h, float* __restrict__ out,
                           long long E, long long Etot)
{
    long long e = (long long)blockIdx.x * 8 + (threadIdx.x >> 5);
    if (e >= Etot) return;
    const int lane = threadIdx.x & 31;
    int s, d;
    if (e < E) { s = src[e]; d = dst[e]; } else { s = d = (int)(e - E); }
    const float alpha = p[e] / z[d];
    const float4 hv = ((const float4*)(h + (size_t)s * 128))[lane];   // b128 gather
    float* __restrict__ orow = out + (size_t)d * 128 + lane * 4;
    atomicAdd(orow + 0, alpha * hv.x);
    atomicAdd(orow + 1, alpha * hv.y);
    atomicAdd(orow + 2, alpha * hv.z);
    atomicAdd(orow + 3, alpha * hv.w);
}

// ---------------- final layer: sigmoid(g2 @ W3 + b3), wave per node ----------------
__global__ __launch_bounds__(256)
void mlp_out_kernel(const _Float16* __restrict__ g2, const float* __restrict__ W3,
                    const float* __restrict__ b3, float* __restrict__ out, int N)
{
    int n = blockIdx.x * 8 + (threadIdx.x >> 5);
    if (n >= N) return;
    int lane = threadIdx.x & 31;
    union { uint4 q; _Float16 h[8]; } rv;
    rv.q = ((const uint4*)(g2 + (size_t)n * 256))[lane];              // 8 halves / lane
    float acc = 0.f;
#pragma unroll
    for (int j = 0; j < 8; ++j) acc += (float)rv.h[j] * W3[lane * 8 + j];
#pragma unroll
    for (int msk = 16; msk; msk >>= 1) acc += __shfl_xor(acc, msk, 32);
    if (lane == 0) {
        float v = acc + b3[0];
        out[n] = 1.0f / (1.0f + __expf(-v));
    }
}

extern "C" void kernel_launch(void* const* d_in, const int* in_sizes, int n_in,
                              void* d_out, int out_size, void* d_ws, size_t ws_size,
                              hipStream_t stream)
{
    const float* x       = (const float*)d_in[0];
    const int*   eidx    = (const int*)  d_in[1];
    const float* W       = (const float*)d_in[2];
    const float* att_src = (const float*)d_in[3];
    const float* att_dst = (const float*)d_in[4];
    const float* bias    = (const float*)d_in[5];
    const float* W1      = (const float*)d_in[6];
    const float* b1      = (const float*)d_in[7];
    const float* W2      = (const float*)d_in[8];
    const float* b2      = (const float*)d_in[9];
    const float* W3      = (const float*)d_in[10];
    const float* b3      = (const float*)d_in[11];
    float* out = (float*)d_out;

    const int N = in_sizes[0] / 128;               // 100000 (multiple of 16)
    const long long E    = (long long)in_sizes[1] / 2;
    const long long Etot = E + N;
    const int* srcIdx = eidx;
    const int* dstIdx = eidx + E;

    // ---- workspace carve-out ----
    char* w = (char*)d_ws;
    float*    hf32   = (float*)w;    w += (size_t)N * 128 * sizeof(float);
    float*    gatf32 = (float*)w;    w += (size_t)N * 128 * sizeof(float);
    _Float16* bufA   = (_Float16*)w; w += (size_t)N * 256 * sizeof(_Float16);
    _Float16* bufB   = (_Float16*)w; w += (size_t)N * 256 * sizeof(_Float16);
    uint4*    Bpack  = (uint4*)w;    w += (size_t)256 * 256 * sizeof(_Float16); // 128KB
    float*    ep     = (float*)w;    w += ((size_t)Etot * sizeof(float) + 255) & ~(size_t)255;
    float*    asrc   = (float*)w;    w += (size_t)N * sizeof(float);
    float*    adst   = (float*)w;    w += (size_t)N * sizeof(float);
    unsigned* mkey   = (unsigned*)w; w += (size_t)N * sizeof(unsigned);
    float*    zsum   = (float*)w;

    const unsigned gemm_grid = (unsigned)((N / 16 + 3) / 4);

    // ---- GAT: h = x @ W (f16 operands, f32 accumulate/output) ----
    cvt_f32_to_f16<<<2048, 256, 0, stream>>>(x, bufA, (long long)N * 32);
    pack_b_kernel<<<(8 * 4 * 32 + 255) / 256, 256, 0, stream>>>(W, Bpack, 128, 128);
    gemm_wmma<128, 128, false, false, false><<<gemm_grid, 128, 0, stream>>>(
        bufA, Bpack, nullptr, hf32, N);

    // ---- attention logits per node ----
    rowdot2_kernel<<<(N + 7) / 8, 256, 0, stream>>>(hf32, att_src, att_dst, asrc, adst, N);

    // ---- edge softmax + aggregation ----
    init_nodes_kernel<<<(N + 255) / 256, 256, 0, stream>>>(mkey, zsum, N);
    init_out_kernel<<<2048, 256, 0, stream>>>(gatf32, bias, (long long)N * 128);

    unsigned egrid = (unsigned)((Etot + 255) / 256);
    edge_logits_kernel<<<egrid, 256, 0, stream>>>(srcIdx, dstIdx, asrc, adst, ep, mkey, E, Etot);
    edge_exp_kernel<<<egrid, 256, 0, stream>>>(dstIdx, ep, mkey, zsum, E, Etot);
    unsigned agrid = (unsigned)((Etot + 7) / 8);
    edge_aggregate_kernel<<<agrid, 256, 0, stream>>>(srcIdx, dstIdx, ep, zsum, hf32, gatf32, E, Etot);

    // ---- MLP layer 1: relu(gat @ W1 + b1) -> f16 ----
    cvt_f32_to_f16<<<2048, 256, 0, stream>>>(gatf32, bufA, (long long)N * 32);
    pack_b_kernel<<<(16 * 4 * 32 + 255) / 256, 256, 0, stream>>>(W1, Bpack, 128, 256);
    gemm_wmma<128, 256, true, true, true><<<gemm_grid, 128, 0, stream>>>(
        bufA, Bpack, b1, bufB, N);

    // ---- MLP layer 2: relu(g1 @ W2 + b2) -> f16 ----
    pack_b_kernel<<<(16 * 8 * 32 + 255) / 256, 256, 0, stream>>>(W2, Bpack, 256, 256);
    gemm_wmma<256, 256, true, true, true><<<gemm_grid, 128, 0, stream>>>(
        bufB, Bpack, b2, bufA, N);

    // ---- MLP layer 3 + sigmoid ----
    mlp_out_kernel<<<(N + 7) / 8, 256, 0, stream>>>(bufA, W3, b3, out, N);
}
